// MultiHeadAttentionFixed_28544352649227
// MI455X (gfx1250) — matrix operand order repaired and checked by reading, not compiled, not run on hardware
//
#include <hip/hip_runtime.h>
#include <hip/hip_bf16.h>

// ---------------------------------------------------------------------------
// Fused causal MHA for B=2, T=2048, D=1024, H=16, HD=64 on gfx1250 (MI455X).
// bf16 WMMA (v_wmma_f32_16x16x32_bf16) with f32 accumulation everywhere.
// All LDS fragment reads are contiguous 16B runs -> ds_load_b128.
// K tiles in the attention kernel are fetched by the Tensor Data Mover
// (tensor_load_to_lds + s_wait_tensorcnt) with hardware LDS row padding.
// Workspace (d_ws): q | k | v | attn_vec, each B*H*T*HD bf16 (8 MB) = 32 MB.
// ---------------------------------------------------------------------------

typedef __bf16 bf16_t;
typedef __attribute__((ext_vector_type(16))) __bf16 v16bf;
typedef __attribute__((ext_vector_type(8)))  float  v8f;
typedef __attribute__((ext_vector_type(4)))  unsigned int u32x4;
typedef __attribute__((ext_vector_type(4)))  int  i32x4;
typedef __attribute__((ext_vector_type(8)))  int  i32x8;

constexpr int Bc = 2, Tc = 2048, Dc = 1024, Hc = 16, HDc = 64;
constexpr int Mtot = Bc * Tc;          // 4096 rows total for projections
constexpr int LDA = 40;                // padded stride for 32-wide bf16 rows (80B, 16B aligned)
constexpr int LDB = 72;                // padded stride for 64-wide bf16 rows (144B, 16B aligned)

#if defined(__has_builtin)
#  if __has_builtin(__builtin_amdgcn_tensor_load_to_lds) && \
      __has_builtin(__builtin_amdgcn_s_wait_tensorcnt)
#    define HAVE_TDM 1
#  endif
#endif
#ifndef HAVE_TDM
#  define HAVE_TDM 0
#endif

__device__ __forceinline__ v8f wmma_bf16(v16bf a, v16bf b, v8f c) {
  return __builtin_amdgcn_wmma_f32_16x16x32_bf16(
      /*neg_a=*/false, a, /*neg_b=*/false, b,
      /*c_mod=*/(short)0, c, /*reuse_a=*/false, /*reuse_b=*/false);
}

// A fragment (16x32, M x K), source row-major [M][stride].
// Contiguous k order used here (same as B): lane<16 -> M=lane, K in {k0+0..15}; lane>=16 -> M=lane-16, K+=16.
// Both 8-element runs are contiguous -> 2 x ds_load_b128.
__device__ __forceinline__ v16bf load_a_frag(const bf16_t* p0, int stride, int k0) {
  const int lane = threadIdx.x & 31;
  const bf16_t* p = p0 + (lane & 15) * stride + k0 + ((lane >> 4) << 4);
  v16bf a;
#pragma unroll
  for (int i = 0; i < 8; ++i) { a[i] = p[i]; a[i + 8] = p[8 + i]; }
  return a;
}

// B fragment (32x16, K x N) where storage is N-major: stored[n][k].
// Contiguous k order used here (same as A): element i -> K = k0 + (lane<16?0:16) + i, N = n0 + lane%16.
// The 16-element run is contiguous -> 2 x ds_load_b128.
__device__ __forceinline__ v16bf load_b_nmajor(const bf16_t* p0, int stride, int k0, int n0) {
  const int lane = threadIdx.x & 31;
  const bf16_t* p = p0 + (n0 + (lane & 15)) * stride + k0 + ((lane >> 4) << 4);
  v16bf b;
#pragma unroll
  for (int i = 0; i < 16; ++i) b[i] = p[i];
  return b;
}

#if HAVE_TDM
// Issue a TDM load of a 64x64 bf16 tile (rows contiguous, row stride HDc
// elements in global) into LDS with 16B padding appended per 128B row
// (-> LDS row stride LDB=72 bf16).  Descriptor per CDNA5 ISA D# layout.
__device__ __forceinline__ void tdm_load_k_tile(const bf16_t* gsrc, bf16_t* lds_dst) {
  const unsigned lds_addr = (unsigned)(unsigned long long)(uintptr_t)lds_dst; // LDS byte offset
  const unsigned long long ga = (unsigned long long)(uintptr_t)gsrc;

  u32x4 g0;
  g0[0] = 1u;                                            // count=1 (valid user D#)
  g0[1] = lds_addr;                                      // lds_addr
  g0[2] = (unsigned)(ga & 0xffffffffu);                  // global_addr[31:0]
  g0[3] = (unsigned)((ga >> 32) & 0x01ffffffu)           // global_addr[56:32]
        | (2u << 30);                                    // type=2 ("image")

  i32x8 g1;
  g1[0] = (1 << 16)                                      // data_size = 2 bytes
        | (1 << 20)                                      // pad_enable
        | (4 << 22)                                      // pad_interval: 32 DWORDs (128B row)
        | (3 << 25);                                     // pad_amount: 4 DWORDs (16B)
  g1[1] = (int)((unsigned)HDc << 16);                    // tensor_dim0 = 64 (low16 in [63:48])
  g1[2] = (int)((unsigned)Tc  << 16);                    // tensor_dim1 = 2048 (low16 in [95:80])
  g1[3] = (64 << 16);                                    // tile_dim0 = 64 ([127:112])
  g1[4] = 64;                                            // tile_dim1 = 64, tile_dim2 = 0
  g1[5] = HDc;                                           // tensor_dim0_stride = 64
  g1[6] = 0;
  g1[7] = 0;

  i32x4 zz = {0, 0, 0, 0};
#if __clang_major__ >= 23
  i32x8 z8 = {};
  __builtin_amdgcn_tensor_load_to_lds(g0, g1, zz, zz, z8, 0);
#else
  __builtin_amdgcn_tensor_load_to_lds(g0, g1, zz, zz, 0);
#endif
}
#endif

// ---------------------------------------------------------------------------
// Kernel 1: fused Q/K/V projection.  grid (Mtot/128, 1024/64, 3), 256 thr.
// z selects weight/bias/output; q gets 1/sqrt(HD)=0.125 scaling fused in.
// Output layout [B, H, T, HD] in bf16.
// ---------------------------------------------------------------------------
__global__ __launch_bounds__(256) void qkv_proj_kernel(
    const float* __restrict__ x,
    const float* __restrict__ wq, const float* __restrict__ bq,
    const float* __restrict__ wk, const float* __restrict__ bk,
    const float* __restrict__ wv, const float* __restrict__ bv,
    bf16_t* __restrict__ qout, bf16_t* __restrict__ kout, bf16_t* __restrict__ vout)
{
  __shared__ __align__(16) bf16_t As[128 * LDA];   // [m][k]   128 x 32 (+pad)
  __shared__ __align__(16) bf16_t Bt[64 * LDA];    // [n][k]   64 x 32 (+pad), transposed

  const int z = blockIdx.z;
  const float* w    = (z == 0) ? wq : (z == 1) ? wk : wv;
  const float* bias = (z == 0) ? bq : (z == 1) ? bk : bv;
  bf16_t* dst       = (z == 0) ? qout : (z == 1) ? kout : vout;
  const float scale = (z == 0) ? 0.125f : 1.0f;

  const int mBase = blockIdx.x * 128;
  const int nBase = blockIdx.y * 64;
  const int tid   = threadIdx.x;
  const int wave  = tid >> 5;
  const int lane  = tid & 31;

  v8f acc[4] = {};

  for (int kb = 0; kb < Dc; kb += 32) {
    __syncthreads();
    { // stage A: 128x32 fp32 -> bf16, row-major
      const int r  = tid >> 1;
      const int c0 = (tid & 1) * 16;
      const float* src = x + (size_t)(mBase + r) * Dc + kb + c0;
      bf16_t* d = As + r * LDA + c0;
#pragma unroll
      for (int j = 0; j < 16; ++j) d[j] = (bf16_t)src[j];
    }
    { // stage B: read w[k][n] coalesced, store transposed Bt[n][k]
      const int r  = tid >> 3;           // k row 0..31
      const int c0 = (tid & 7) * 8;      // n col group
      const float* src = w + (size_t)(kb + r) * Dc + nBase + c0;
#pragma unroll
      for (int j = 0; j < 8; ++j) Bt[(c0 + j) * LDA + r] = (bf16_t)src[j];
    }
    if (kb + 32 < Dc) { // prefetch next k-block (global_prefetch_b8)
      const int r = tid >> 1;
      __builtin_prefetch(x + (size_t)(mBase + r) * Dc + kb + 32, 0, 1);
      __builtin_prefetch(w + (size_t)(kb + 32 + (tid & 31)) * Dc + nBase, 0, 1);
    }
    __syncthreads();

    v16bf a = load_a_frag(As + (wave * 16) * LDA, LDA, 0);
#pragma unroll
    for (int nt = 0; nt < 4; ++nt) {
      v16bf b = load_b_nmajor(Bt, LDA, 0, nt * 16);
      acc[nt] = wmma_bf16(a, b, acc[nt]);
    }
  }

  // epilogue: bias + scale, scatter to [B,H,T,HD] bf16
  const int nloc = lane & 15;
  const int rofs = (lane < 16) ? 0 : 8;
  const int h    = nBase >> 6;   // 64-wide tile == one head
#pragma unroll
  for (int nt = 0; nt < 4; ++nt) {
    const int col = nBase + nt * 16 + nloc;
    const float bsc = bias[col];
    const int hd = col & 63;
#pragma unroll
    for (int r = 0; r < 8; ++r) {
      const int row = mBase + wave * 16 + rofs + r;
      const int bb  = row >> 11;          // / 2048
      const int tt  = row & 2047;
      const float val = (acc[nt][r] + bsc) * scale;
      dst[(((size_t)bb * Hc + h) * Tc + tt) * HDc + hd] = (bf16_t)val;
    }
  }
}

// ---------------------------------------------------------------------------
// Kernel 2: causal flash attention per (64-row tile, head, batch).
// 128 threads = 4 waves; wave w owns query rows [qRow0, qRow0+16).
// K tile: TDM DMA into LDS (row-major [kv][hd], padded rows).
// V tile: manual transposed stage [hd][kv] (PV B-frags contiguous).
// ---------------------------------------------------------------------------
__global__ __launch_bounds__(128) void attn_kernel(
    const bf16_t* __restrict__ qm, const bf16_t* __restrict__ km,
    const bf16_t* __restrict__ vm, bf16_t* __restrict__ aout)
{
  __shared__ __align__(16) bf16_t Kt [64 * LDB];   // [kv][hd]
  __shared__ __align__(16) bf16_t VtT[64 * LDB];   // [hd][kv]
  __shared__ __align__(16) bf16_t Pt [64 * LDB];   // per-wave 16x64 P staging

  const int rowTile = blockIdx.x;     // 0..31
  const int h       = blockIdx.y;
  const int b       = blockIdx.z;
  const int tid  = threadIdx.x;
  const int wave = tid >> 5;
  const int lane = tid & 31;
  const size_t headOff = ((size_t)b * Hc + h) * (size_t)Tc * HDc;

  const int qRow0 = rowTile * 64 + wave * 16;
  const bf16_t* qp = qm + headOff + (size_t)qRow0 * HDc;
  const v16bf aq0 = load_a_frag(qp, HDc, 0);    // Q already scaled by 1/8
  const v16bf aq1 = load_a_frag(qp, HDc, 32);

  float mi[8], li[8];
  v8f o[4] = {};
#pragma unroll
  for (int r = 0; r < 8; ++r) { mi[r] = -1e30f; li[r] = 0.f; }

  const int nloc = lane & 15;
  const int rofs = (lane < 16) ? 0 : 8;

  for (int j = 0; j <= rowTile; ++j) {
    __syncthreads();

#if HAVE_TDM
    if (wave == 0) {   // single wave issues the K-tile DMA (EXEC ignored by TDM)
      tdm_load_k_tile(km + headOff + (size_t)(j * 64) * HDc, Kt);
    }
#else
    { // fallback: manual K stage, 16B vector copies
      const int r  = tid >> 1;
      const int c0 = (tid & 1) * 32;
      const uint4* ksrc = (const uint4*)(km + headOff + (size_t)(j * 64 + r) * HDc + c0);
      uint4* kd = (uint4*)(Kt + r * LDB + c0);
#pragma unroll
      for (int u = 0; u < 4; ++u) kd[u] = ksrc[u];
    }
#endif

    { // stage V transposed: coalesced global read, scattered LDS write
      const int r  = tid >> 1;           // kv row 0..63
      const int c0 = (tid & 1) * 32;     // hd half
      uint4 vv[4];
      const uint4* vsrc = (const uint4*)(vm + headOff + (size_t)(j * 64 + r) * HDc + c0);
#pragma unroll
      for (int u = 0; u < 4; ++u) vv[u] = vsrc[u];
      const bf16_t* ve = (const bf16_t*)vv;
#pragma unroll
      for (int u = 0; u < 32; ++u) VtT[(c0 + u) * LDB + r] = ve[u];  // [hd][kv]
    }
    if (j < rowTile) { // prefetch next V tile (global_prefetch_b8)
      const int r = tid >> 1;
      __builtin_prefetch(vm + headOff + (size_t)((j + 1) * 64 + r) * HDc, 0, 1);
    }

#if HAVE_TDM
    if (wave == 0) __builtin_amdgcn_s_wait_tensorcnt(0);  // K tile landed
#endif
    __syncthreads();

    // S = Q * K^T   (16x64 per wave, HD=64 -> two k-slices)
    v8f s[4] = {};
#pragma unroll
    for (int nt = 0; nt < 4; ++nt) {
      v16bf b0 = load_b_nmajor(Kt, LDB, 0,  nt * 16);   // B[d][n] = Kt[n][d]
      s[nt] = wmma_bf16(aq0, b0, s[nt]);
      v16bf b1 = load_b_nmajor(Kt, LDB, 32, nt * 16);
      s[nt] = wmma_bf16(aq1, b1, s[nt]);
    }

    if (j == rowTile) { // causal mask, diagonal tile only
#pragma unroll
      for (int nt = 0; nt < 4; ++nt) {
        const int col = j * 64 + nt * 16 + nloc;
#pragma unroll
        for (int r = 0; r < 8; ++r) {
          const int row = qRow0 + rofs + r;
          if (col > row) s[nt][r] = -1e30f;
        }
      }
    }

    // online softmax: rows 0-7 live in lanes 0-15, rows 8-15 in lanes 16-31
    float mnew[8], corr[8], rsum[8];
#pragma unroll
    for (int r = 0; r < 8; ++r) {
      float mx = fmaxf(fmaxf(s[0][r], s[1][r]), fmaxf(s[2][r], s[3][r]));
#pragma unroll
      for (int m = 1; m < 16; m <<= 1) mx = fmaxf(mx, __shfl_xor(mx, m, 32));
      mnew[r] = fmaxf(mi[r], mx);
      corr[r] = __expf(mi[r] - mnew[r]);
      rsum[r] = 0.f;
    }

    bf16_t* pw = Pt + wave * 16 * LDB;   // wave-private P region
#pragma unroll
    for (int nt = 0; nt < 4; ++nt) {
#pragma unroll
      for (int r = 0; r < 8; ++r) {
        const float p = __expf(s[nt][r] - mnew[r]);
        rsum[r] += p;
        pw[(rofs + r) * LDB + nt * 16 + nloc] = (bf16_t)p;   // D-layout -> LDS
      }
    }
#pragma unroll
    for (int r = 0; r < 8; ++r) {
      float rs = rsum[r];
#pragma unroll
      for (int m = 1; m < 16; m <<= 1) rs += __shfl_xor(rs, m, 32);
      li[r] = li[r] * corr[r] + rs;
      mi[r] = mnew[r];
    }
#pragma unroll
    for (int nt = 0; nt < 4; ++nt)
#pragma unroll
      for (int r = 0; r < 8; ++r) o[nt][r] *= corr[r];

    __syncthreads();   // make P visible before A-layout reload

    // O += P * V  (B-frags contiguous from transposed V tile)
    const v16bf ap0 = load_a_frag(pw, LDB, 0);
    const v16bf ap1 = load_a_frag(pw, LDB, 32);
#pragma unroll
    for (int nt = 0; nt < 4; ++nt) {
      v16bf b0 = load_b_nmajor(VtT, LDB, 0,  nt * 16);  // B[kv][hd] = VtT[hd][kv]
      o[nt] = wmma_bf16(ap0, b0, o[nt]);
      v16bf b1 = load_b_nmajor(VtT, LDB, 32, nt * 16);
      o[nt] = wmma_bf16(ap1, b1, o[nt]);
    }
  }

  // normalize and store attn_vec bf16 [B*T, 1024]
#pragma unroll
  for (int nt = 0; nt < 4; ++nt) {
#pragma unroll
    for (int r = 0; r < 8; ++r) {
      const int row = qRow0 + rofs + r;
      const float val = o[nt][r] / li[r];
      aout[((size_t)b * Tc + row) * Dc + h * HDc + nt * 16 + nloc] = (bf16_t)val;
    }
  }
}

// ---------------------------------------------------------------------------
// Kernel 3: output projection  attn_vec(bf16) @ wo + bo -> fp32 d_out.
// ---------------------------------------------------------------------------
__global__ __launch_bounds__(256) void out_proj_kernel(
    const bf16_t* __restrict__ av,
    const float* __restrict__ wo, const float* __restrict__ bo,
    float* __restrict__ out)
{
  __shared__ __align__(16) bf16_t As[128 * LDA];   // [m][k]
  __shared__ __align__(16) bf16_t Bt[64 * LDA];    // [n][k] transposed

  const int mBase = blockIdx.x * 128;
  const int nBase = blockIdx.y * 64;
  const int tid  = threadIdx.x;
  const int wave = tid >> 5;
  const int lane = tid & 31;

  v8f acc[4] = {};

  for (int kb = 0; kb < Dc; kb += 32) {
    __syncthreads();
    { // A already bf16: 16B vector copies
      const int r  = tid >> 1;
      const int c0 = (tid & 1) * 16;
      const uint4* src = (const uint4*)(av + (size_t)(mBase + r) * Dc + kb + c0);
      uint4* d = (uint4*)(As + r * LDA + c0);
      d[0] = src[0]; d[1] = src[1];
    }
    { // B: read wo[k][n] coalesced, store transposed Bt[n][k]
      const int r  = tid >> 3;
      const int c0 = (tid & 7) * 8;
      const float* src = wo + (size_t)(kb + r) * Dc + nBase + c0;
#pragma unroll
      for (int j = 0; j < 8; ++j) Bt[(c0 + j) * LDA + r] = (bf16_t)src[j];
    }
    if (kb + 32 < Dc) {
      const int r = tid >> 1;
      __builtin_prefetch(av + (size_t)(mBase + r) * Dc + kb + 32, 0, 1);
      __builtin_prefetch(wo + (size_t)(kb + 32 + (tid & 31)) * Dc + nBase, 0, 1);
    }
    __syncthreads();

    v16bf a = load_a_frag(As + (wave * 16) * LDA, LDA, 0);
#pragma unroll
    for (int nt = 0; nt < 4; ++nt) {
      v16bf b = load_b_nmajor(Bt, LDA, 0, nt * 16);
      acc[nt] = wmma_bf16(a, b, acc[nt]);
    }
  }

  const int nloc = lane & 15;
  const int rofs = (lane < 16) ? 0 : 8;
#pragma unroll
  for (int nt = 0; nt < 4; ++nt) {
    const int col = nBase + nt * 16 + nloc;
    const float bsc = bo[col];
#pragma unroll
    for (int r = 0; r < 8; ++r) {
      const int row = mBase + wave * 16 + rofs + r;
      out[(size_t)row * Dc + col] = acc[nt][r] + bsc;
    }
  }
}

// ---------------------------------------------------------------------------
extern "C" void kernel_launch(void* const* d_in, const int* in_sizes, int n_in,
                              void* d_out, int out_size, void* d_ws, size_t ws_size,
                              hipStream_t stream) {
  const float* x  = (const float*)d_in[0];
  const float* wq = (const float*)d_in[1];
  const float* bq = (const float*)d_in[2];
  const float* wk = (const float*)d_in[3];
  const float* bk = (const float*)d_in[4];
  const float* wv = (const float*)d_in[5];
  const float* bv = (const float*)d_in[6];
  const float* wo = (const float*)d_in[7];
  const float* bo = (const float*)d_in[8];
  float* out = (float*)d_out;

  const size_t per = (size_t)Bc * Hc * Tc * HDc;   // 4M bf16 elements each
  bf16_t* qb = (bf16_t*)d_ws;
  bf16_t* kb = qb + per;
  bf16_t* vb = kb + per;
  bf16_t* ab = vb + per;                           // attn_vec [B*T, 1024]

  dim3 g1(Mtot / 128, (Hc * HDc) / 64, 3);
  qkv_proj_kernel<<<g1, 256, 0, stream>>>(x, wq, bq, wk, bk, wv, bv, qb, kb, vb);

  dim3 g2(Tc / 64, Hc, Bc);
  attn_kernel<<<g2, 128, 0, stream>>>(qb, kb, vb, ab);

  dim3 g3(Mtot / 128, Dc / 64);
  out_proj_kernel<<<g3, 256, 0, stream>>>(ab, wo, bo, out);
}
